// VectorQuantizer_28106265985618
// MI455X (gfx1250) — matrix operand order repaired and checked by reading, not compiled, not run on hardware
//
#include <hip/hip_runtime.h>
#include <hip/hip_bf16.h>

#define NUM_E   512
#define DIM     100
#define K_PAD   136          // padded f16 row stride: 272B -> conflict-free b128 LDS reads
#define ROWS_PB 128          // rows per block
#define THREADS 256          // 8 wave32
#define N_ROWS  (64*64*64)   // 262144
#define N_BLOCKS (N_ROWS / ROWS_PB) // 2048
#define TOTAL_ELEMS (N_ROWS * DIM)  // 26214400

typedef _Float16 v16h __attribute__((ext_vector_type(16)));
typedef _Float16 v8h  __attribute__((ext_vector_type(8)));
typedef float    v8f  __attribute__((ext_vector_type(8)));

// ---------------- kernel 0: codebook -> f16 padded [512][136] + ||e||^2 ----------------
__global__ void vq_prep(const float* __restrict__ emb,
                        _Float16* __restrict__ emb16,
                        float* __restrict__ norms) {
    int row = blockIdx.x * blockDim.x + threadIdx.x;
    if (row >= NUM_E) return;
    float s = 0.f;
    for (int k = 0; k < DIM; ++k) {
        float v = emb[row * DIM + k];
        s += v * v;
        emb16[row * K_PAD + k] = (_Float16)v;
    }
    for (int k = DIM; k < K_PAD; ++k) emb16[row * K_PAD + k] = (_Float16)0.f;
    norms[row] = s;
}

// ---------------- kernel 1: fused distances + argmin + gather + loss partials ----------------
__global__ void __launch_bounds__(THREADS, 1)
vq_main(const float* __restrict__ x,
        const float* __restrict__ emb,
        const _Float16* __restrict__ emb16,
        const float* __restrict__ norms,
        float* __restrict__ out,
        float* __restrict__ partials) {
    extern __shared__ char smem[];
    _Float16* sEmb  = (_Float16*)smem;                                    // 512*136 f16
    _Float16* sIn   = (_Float16*)(smem + NUM_E * K_PAD * 2);              // 128*136 f16
    float*    sNorm = (float*)(smem + (NUM_E + ROWS_PB) * K_PAD * 2);     // 512 f32
    float*    sRed  = sNorm + NUM_E;                                      // 8 f32

    const int tid  = threadIdx.x;
    const int wave = tid >> 5;
    const int lane = tid & 31;
    const int lhalf = lane >> 4;   // 0: lanes 0-15, 1: lanes 16-31
    const int l15   = lane & 15;

    // ---- stage f16 codebook into LDS via CDNA5 async global->LDS DMA (ASYNCcnt) ----
    {
        const uint4* src = (const uint4*)emb16;
        uint4* dst = (uint4*)sEmb;
        const int n16 = NUM_E * K_PAD * 2 / 16;   // 8704 x 16B
        for (int i = tid; i < n16; i += THREADS) {
            unsigned ldsOff = (unsigned)(uintptr_t)(dst + i);  // low 32 bits == LDS offset
            const uint4* g = src + i;
            asm volatile("global_load_async_to_lds_b128 %0, %1, off"
                         :: "v"(ldsOff), "v"(g) : "memory");
        }
        for (int i = tid; i < NUM_E; i += THREADS) sNorm[i] = norms[i];
    }

    // stage 128-row input tile, f32 -> f16, zero-pad K to 128 (overlaps with async DMA)
    const long rowBase = (long)blockIdx.x * ROWS_PB;
    {
        const float* xin = x + rowBase * DIM;
        for (int i = tid; i < ROWS_PB * DIM; i += THREADS) {
            int r = i / DIM, c = i - r * DIM;
            sIn[r * K_PAD + c] = (_Float16)xin[i];
        }
        for (int i = tid; i < ROWS_PB * 28; i += THREADS) {  // cols 100..127
            int r = i / 28, c = DIM + (i - r * 28);
            sIn[r * K_PAD + c] = (_Float16)0.f;
        }
    }
    asm volatile("s_wait_asynccnt 0" ::: "memory");
    __syncthreads();

    // ---- per-wave: 16 rows, sweep 32 column tiles of the codebook ----
    const int rloc = wave * 16;

    // A fragments (16x32 f16 per K-tile); contiguous K layout (matches B):
    //   lanes 0-15 : M=lane,    halfs[0..15]=K+0..15
    //   lanes 16-31: M=lane-16, halfs[0..15]=K+16..31
    v16h afrag[4];
#pragma unroll
    for (int kt = 0; kt < 4; ++kt) {
        const _Float16* pa = sIn + (rloc + l15) * K_PAD + kt * 32 + lhalf * 16;
        v8h lo = *(const v8h*)(pa);
        v8h hi = *(const v8h*)(pa + 8);
        afrag[kt] = __builtin_shufflevector(lo, hi, 0,1,2,3,4,5,6,7,8,9,10,11,12,13,14,15);
    }

    float best[8];
    int   bidx[8];
#pragma unroll
    for (int v = 0; v < 8; ++v) { best[v] = -3.0e38f; bidx[v] = 0; }

    // Unroll x2: two independent WMMA accumulator chains so tile n+1's WMMAs
    // co-execute with tile n's compare/select VALU (hides the F16-WMMA->VALU
    // 4-slot hazard that otherwise becomes v_nops).
    for (int n = 0; n < 32; n += 2) {
        float bias0 = -0.5f * sNorm[(n + 0) * 16 + l15];
        float bias1 = -0.5f * sNorm[(n + 1) * 16 + l15];
        v8f c0, c1;
#pragma unroll
        for (int v = 0; v < 8; ++v) { c0[v] = bias0; c1[v] = bias1; }

#pragma unroll
        for (int kt = 0; kt < 4; ++kt) {
            const _Float16* pb = sEmb + ((n + 0) * 16 + l15) * K_PAD + kt * 32 + lhalf * 16;
            v8h blo = *(const v8h*)(pb);
            v8h bhi = *(const v8h*)(pb + 8);
            v16h bfr = __builtin_shufflevector(blo, bhi, 0,1,2,3,4,5,6,7,8,9,10,11,12,13,14,15);
            c0 = __builtin_amdgcn_wmma_f32_16x16x32_f16(
                     false, afrag[kt], false, bfr, (short)0, c0, false, false);
        }
#pragma unroll
        for (int kt = 0; kt < 4; ++kt) {
            const _Float16* pb = sEmb + ((n + 1) * 16 + l15) * K_PAD + kt * 32 + lhalf * 16;
            v8h blo = *(const v8h*)(pb);
            v8h bhi = *(const v8h*)(pb + 8);
            v16h bfr = __builtin_shufflevector(blo, bhi, 0,1,2,3,4,5,6,7,8,9,10,11,12,13,14,15);
            c1 = __builtin_amdgcn_wmma_f32_16x16x32_f16(
                     false, afrag[kt], false, bfr, (short)0, c1, false, false);
        }

        const int idx0 = (n + 0) * 16 + l15;
        const int idx1 = (n + 1) * 16 + l15;
#pragma unroll
        for (int v = 0; v < 8; ++v)
            if (c0[v] > best[v]) { best[v] = c0[v]; bidx[v] = idx0; }
#pragma unroll
        for (int v = 0; v < 8; ++v)
            if (c1[v] > best[v]) { best[v] = c1[v]; bidx[v] = idx1; }
    }

    // cross-lane argmax within each 16-lane half (xor 1,2,4,8 stays inside the half)
#pragma unroll
    for (int v = 0; v < 8; ++v) {
        float b = best[v]; int bi = bidx[v];
#pragma unroll
        for (int off = 1; off < 16; off <<= 1) {
            float ob = __shfl_xor(b, off, 32);
            int   oi = __shfl_xor(bi, off, 32);
            if (ob > b || (ob == b && oi < bi)) { b = ob; bi = oi; }
        }
        best[v] = b; bidx[v] = bi;
    }

    // ---- gather f32 codebook rows, write output, accumulate (q-x)^2 ----
    float lsum = 0.f;
#pragma unroll
    for (int v = 0; v < 8; ++v) {
        long row = rowBase + rloc + v + lhalf * 8;   // half-wave owns 8 rows
        int  e   = bidx[v];                          // uniform across the half
        for (int col = l15; col < DIM; col += 16) {  // 16 lanes -> coalesced 64B
            float q  = emb[e * DIM + col];
            float xi = x[row * DIM + col];
            out[row * DIM + col] = q;
            float d = q - xi;
            lsum += d * d;
        }
    }

    // deterministic block reduction of loss partial
#pragma unroll
    for (int off = 16; off > 0; off >>= 1) lsum += __shfl_xor(lsum, off, 32);
    if (lane == 0) sRed[wave] = lsum;
    __syncthreads();
    if (tid == 0) {
        float t = 0.f;
        for (int w = 0; w < 8; ++w) t += sRed[w];
        partials[blockIdx.x] = t;
    }
}

// ---------------- kernel 2: reduce partials -> the two scalar losses ----------------
__global__ void vq_loss(const float* __restrict__ partials, float* __restrict__ out2) {
    __shared__ float red[8];
    float s = 0.f;
    for (int i = threadIdx.x; i < N_BLOCKS; i += blockDim.x) s += partials[i];
#pragma unroll
    for (int off = 16; off > 0; off >>= 1) s += __shfl_xor(s, off, 32);
    int wave = threadIdx.x >> 5, lane = threadIdx.x & 31;
    if (lane == 0) red[wave] = s;
    __syncthreads();
    if (threadIdx.x == 0) {
        float t = 0.f;
        for (int w = 0; w < 8; ++w) t += red[w];
        float mean = t / (float)TOTAL_ELEMS;
        out2[0] = mean;          // quantization_loss
        out2[1] = 0.25f * mean;  // commitment_loss
    }
}

extern "C" void kernel_launch(void* const* d_in, const int* in_sizes, int n_in,
                              void* d_out, int out_size, void* d_ws, size_t ws_size,
                              hipStream_t stream) {
    const float* x   = (const float*)d_in[0];   // inputs  [64,64,64,100]
    const float* emb = (const float*)d_in[1];   // embedding [512,100]

    // workspace layout
    _Float16* emb16    = (_Float16*)d_ws;                              // 512*136 f16 = 139264 B
    float*    norms    = (float*)((char*)d_ws + NUM_E * K_PAD * 2);    // 512 f32
    float*    partials = norms + NUM_E;                                // 2048 f32

    float* out = (float*)d_out;

    vq_prep<<<(NUM_E + 255) / 256, 256, 0, stream>>>(emb, emb16, norms);

    const size_t smemBytes = (size_t)(NUM_E + ROWS_PB) * K_PAD * 2 + NUM_E * 4 + 8 * 4;
    (void)hipFuncSetAttribute((const void*)vq_main,
                              hipFuncAttributeMaxDynamicSharedMemorySize,
                              (int)smemBytes);
    vq_main<<<N_BLOCKS, THREADS, smemBytes, stream>>>(x, emb, emb16, norms, out, partials);

    vq_loss<<<1, 256, 0, stream>>>(partials, out + (long)TOTAL_ELEMS);
}
